// Net_70832600646051
// MI455X (gfx1250) — matrix operand order repaired and checked by reading, not compiled, not run on hardware
//
#include <hip/hip_runtime.h>
#include <hip/hip_bf16.h>

#define N_NODES   100000
#define N_EDGES   3200000
#define F_IN      1024
#define HIDDEN    64
#define N_CLASSES 32

typedef __attribute__((ext_vector_type(16))) _Float16 v16h;
typedef __attribute__((ext_vector_type(8)))  float    v8f;

union Frag16 { v16h v; unsigned u[8]; };

__device__ inline unsigned pack2h(float a, float b) {
    union { _Float16 h[2]; unsigned u; } p;
    p.h[0] = (_Float16)a;
    p.h[1] = (_Float16)b;
    return p.u;
}

// Truncating a generic pointer to a __shared__ object yields the LDS byte
// offset (generic LDS addr = SHARED_BASE aperture in [63:32] | offset in [31:0]).
__device__ inline unsigned lds_addr(const void* p) {
    return (unsigned)(size_t)p;
}

// ---------------------------------------------------------------------------
// Convert W1 [64,1024] and W2 [32,64] (row-major f32) to packed-f16 pairs.
// ---------------------------------------------------------------------------
__global__ void convert_weights(const float* __restrict__ W1,
                                const float* __restrict__ W2,
                                unsigned* __restrict__ w1p,
                                unsigned* __restrict__ w2p) {
    int t = blockIdx.x * blockDim.x + threadIdx.x;     // packed-pair index
    if (t < (HIDDEN * F_IN) / 2) {                     // 32768 pairs
        w1p[t] = pack2h(W1[2 * t], W1[2 * t + 1]);
    } else {
        int s = t - (HIDDEN * F_IN) / 2;
        if (s < (N_CLASSES * HIDDEN) / 2)              // 1024 pairs
            w2p[s] = pack2h(W2[2 * s], W2[2 * s + 1]);
    }
}

// ---------------------------------------------------------------------------
// GEMM1: h1[N,64] = x[N,1024] @ W1^T   via v_wmma_f32_16x16x32_f16.
// 256 threads = 8 waves; M-tile 64, N = 64 (full HIDDEN).
// Wave w: M-subtile (w&3), N-half (w>>2)*32 -> two WMMA per K-step.
//
// x (409.6 MB, read once -> ~17.6us floor at 23.3 TB/s) is streamed into a
// double-buffered LDS tile with GLOBAL_LOAD_ASYNC_TO_LDS_B128 (ASYNCcnt):
// next tile's DMA is issued before this tile's WMMAs, wait+barrier after, so
// HBM latency overlaps matrix work with a single barrier per K-step.
// f32->f16 conversion happens in the fragment build (VALU co-executes with
// the XDL WMMA pipe). Row index is clamped instead of predicated: only the
// last block over-reads (row 99999 duplicates), and stores are guarded.
// ---------------------------------------------------------------------------
__global__ __launch_bounds__(256)
void gemm1_wmma(const float* __restrict__ x,
                const unsigned* __restrict__ w1p,
                float* __restrict__ h1) {
    __shared__ float As[2][64 * 32];   // 2 x 8KB f32 tiles (64 rows x 32 K)

    const int t    = threadIdx.x;
    const int lane = t & 31;
    const int wave = t >> 5;
    const int m0   = blockIdx.x * 64;
    const int wm   = wave & 3;         // M subtile 0..3
    const int wn   = wave >> 2;        // N base = wn*32
    const int li   = lane & 15;
    const int hi   = lane >> 4;

    // Issue one 64x32 f32 tile (8KB) as 512 async b128 transfers, 2/thread.
    auto issue_tile = [&](int kb, int buf) {
        #pragma unroll
        for (int i = 0; i < 2; ++i) {
            int c  = t + 256 * i;          // 0..511
            int m  = c >> 3;               // 8 x b128 chunks per row
            int k4 = (c & 7) << 2;         // f32 column base
            int row = m0 + m;
            row = row < N_NODES ? row : (N_NODES - 1);   // clamp (see above)
            const float* g = x + (size_t)row * F_IN + kb + k4;
            unsigned l = lds_addr(&As[buf][m * 32 + k4]);
            asm volatile("global_load_async_to_lds_b128 %0, %1, off"
                         :: "v"(l), "v"(g) : "memory");
        }
    };

    v8f acc0 = {};
    v8f acc1 = {};

    issue_tile(0, 0);
    asm volatile("s_wait_asynccnt 0x0" ::: "memory");
    __syncthreads();

    for (int kt = 0; kt < F_IN / 32; ++kt) {
        const int kb  = kt * 32;
        const int cur = kt & 1;

        if (kt + 1 < F_IN / 32)
            issue_tile(kb + 32, cur ^ 1);   // prefetch next tile (async DMA)

        // A fragment (contiguous K order, matching B): lane(li,hi) needs
        // K in [16*hi, 16*hi+16) of row wm*16+li.
        const float* ap = &As[cur][(wm * 16 + li) * 32];
        float4 r0 = *(const float4*)(ap + 16 * hi);
        float4 r1 = *(const float4*)(ap + 16 * hi + 4);
        float4 r2 = *(const float4*)(ap + 16 * hi + 8);
        float4 r3 = *(const float4*)(ap + 16 * hi + 12);
        Frag16 a;
        a.u[0] = pack2h(r0.x, r0.y); a.u[1] = pack2h(r0.z, r0.w);
        a.u[2] = pack2h(r1.x, r1.y); a.u[3] = pack2h(r1.z, r1.w);
        a.u[4] = pack2h(r2.x, r2.y); a.u[5] = pack2h(r2.z, r2.w);
        a.u[6] = pack2h(r3.x, r3.y); a.u[7] = pack2h(r3.z, r3.w);

        // B fragments (32x16): lane = column n, dword j -> K = kb+2j+16*hi,
        // i.e. pair (kb>>1)+j+8*hi of W1 row n (contiguous -> b128 loads).
        Frag16 b0, b1;
        #pragma unroll
        for (int j = 0; j < 8; ++j) {
            int kp = (kb >> 1) + j + 8 * hi;
            int n0 = wn * 32 + li;
            int n1 = wn * 32 + 16 + li;
            b0.u[j] = w1p[n0 * (F_IN / 2) + kp];
            b1.u[j] = w1p[n1 * (F_IN / 2) + kp];
        }

        acc0 = __builtin_amdgcn_wmma_f32_16x16x32_f16(
            false, a.v, false, b0.v, (short)0, acc0, false, false);
        acc1 = __builtin_amdgcn_wmma_f32_16x16x32_f16(
            false, a.v, false, b1.v, (short)0, acc1, false, false);

        asm volatile("s_wait_asynccnt 0x0" ::: "memory");
        __syncthreads();
    }

    // C/D layout: VGPR i -> row M = i + 8*hi; lane li -> column
    #pragma unroll
    for (int i = 0; i < 8; ++i) {
        int row = m0 + wm * 16 + i + 8 * hi;
        if (row < N_NODES) {
            h1[(size_t)row * HIDDEN + wn * 32 + li]      = acc0[i];
            h1[(size_t)row * HIDDEN + wn * 32 + 16 + li] = acc1[i];
        }
    }
}

// ---------------------------------------------------------------------------
// Scatter layer 1: agg1[dst] += h1[src] (edge-parallel; h1/agg1 are 25.6 MB
// each -> fully L2-resident atomics on the 192 MB L2).
// ---------------------------------------------------------------------------
__global__ __launch_bounds__(256)
void scatter1(const float* __restrict__ h1,
              const int* __restrict__ ei,
              float* __restrict__ agg) {
    int gid = blockIdx.x * blockDim.x + threadIdx.x;
    int e = gid >> 6;
    int f = gid & 63;
    if (e < N_EDGES) {
        int s = ei[e];
        int d = ei[N_EDGES + e];
        atomicAdd(&agg[(size_t)d * HIDDEN + f], h1[(size_t)s * HIDDEN + f]);
    }
}

// ---------------------------------------------------------------------------
// GEMM2: h2[N,32] = relu(agg1 + b1) @ W2^T, bias+ReLU fused into A staging.
// K = 64 -> two WMMA steps; wave w: M-subtile (w&3), N-subtile (w>>2)*16.
// Row clamp instead of per-lane predication (stores are guarded).
// ---------------------------------------------------------------------------
__global__ __launch_bounds__(256)
void gemm2_wmma(const float* __restrict__ agg,
                const float* __restrict__ b1,
                const unsigned* __restrict__ w2p,
                float* __restrict__ h2) {
    __shared__ unsigned As[64 * 32];   // 64 rows x 32 k-pairs (64 K) = 8KB

    const int t    = threadIdx.x;
    const int lane = t & 31;
    const int wave = t >> 5;
    const int m0   = blockIdx.x * 64;
    const int wm   = wave & 3;
    const int wn   = wave >> 2;        // N base = wn*16
    const int li   = lane & 15;
    const int hi   = lane >> 4;

    #pragma unroll
    for (int i = 0; i < 8; ++i) {
        int idx = t + 256 * i;         // 0..2047
        int m   = idx >> 5;
        int kp  = idx & 31;
        int row = m0 + m;
        row = row < N_NODES ? row : (N_NODES - 1);
        const float* p = agg + (size_t)row * HIDDEN + 2 * kp;
        float a0 = fmaxf(p[0] + b1[2 * kp],     0.f);
        float a1 = fmaxf(p[1] + b1[2 * kp + 1], 0.f);
        As[idx] = pack2h(a0, a1);
    }
    __syncthreads();

    v8f acc = {};
    #pragma unroll
    for (int kt = 0; kt < 2; ++kt) {
        const int kb = kt * 32;
        Frag16 a, b;
        #pragma unroll
        for (int j = 0; j < 8; ++j) {
            int kp = (kb >> 1) + j + 8 * hi;
            a.u[j] = As[(wm * 16 + li) * 32 + kp];
        }
        #pragma unroll
        for (int j = 0; j < 8; ++j) {
            int kp = (kb >> 1) + j + 8 * hi;
            int n  = wn * 16 + li;
            b.u[j] = w2p[n * (HIDDEN / 2) + kp];
        }
        acc = __builtin_amdgcn_wmma_f32_16x16x32_f16(
            false, a.v, false, b.v, (short)0, acc, false, false);
    }

    #pragma unroll
    for (int i = 0; i < 8; ++i) {
        int row = m0 + wm * 16 + i + 8 * hi;
        if (row < N_NODES)
            h2[(size_t)row * N_CLASSES + wn * 16 + li] = acc[i];
    }
}

// ---------------------------------------------------------------------------
// out = b2 (broadcast), then scatter layer 2: out[dst] += h2[src]
// ---------------------------------------------------------------------------
__global__ void init_out(const float* __restrict__ b2, float* __restrict__ out) {
    int gid = blockIdx.x * blockDim.x + threadIdx.x;
    if (gid < N_NODES * N_CLASSES) out[gid] = b2[gid & (N_CLASSES - 1)];
}

__global__ __launch_bounds__(256)
void scatter2(const float* __restrict__ h2,
              const int* __restrict__ ei,
              float* __restrict__ out) {
    int gid = blockIdx.x * blockDim.x + threadIdx.x;
    int e = gid >> 5;
    int f = gid & 31;
    if (e < N_EDGES) {
        int s = ei[e];
        int d = ei[N_EDGES + e];
        atomicAdd(&out[(size_t)d * N_CLASSES + f], h2[(size_t)s * N_CLASSES + f]);
    }
}

// ---------------------------------------------------------------------------
// Workspace layout (needs ~51.3 MB):
//   [0)         w1p  packed f16 W1         131072 B
//   [131072)    w2p  packed f16 W2           4096 B
//   [135168)    hbuf h1 (reused as h2)      N*64*4 = 25,600,000 B
//   [+25.6MB)   agg1                        N*64*4 = 25,600,000 B
// ---------------------------------------------------------------------------
extern "C" void kernel_launch(void* const* d_in, const int* in_sizes, int n_in,
                              void* d_out, int out_size, void* d_ws, size_t ws_size,
                              hipStream_t stream) {
    const float* x  = (const float*)d_in[0];
    const float* W1 = (const float*)d_in[1];
    const float* b1 = (const float*)d_in[2];
    const float* W2 = (const float*)d_in[3];
    const float* b2 = (const float*)d_in[4];
    const int*   ei = (const int*)d_in[5];
    float* out = (float*)d_out;

    char* ws = (char*)d_ws;
    unsigned* w1p = (unsigned*)ws;
    unsigned* w2p = (unsigned*)(ws + 131072);
    float* hbuf   = (float*)(ws + 131072 + 4096);
    float* agg1   = (float*)(ws + 131072 + 4096 + (size_t)N_NODES * HIDDEN * 4);

    hipMemsetAsync(agg1, 0, (size_t)N_NODES * HIDDEN * 4, stream);

    convert_weights<<<((HIDDEN * F_IN + N_CLASSES * HIDDEN) / 2 + 255) / 256,
                      256, 0, stream>>>(W1, W2, w1p, w2p);

    const int mblocks = (N_NODES + 63) / 64;   // 1563
    gemm1_wmma<<<mblocks, 256, 0, stream>>>(x, w1p, hbuf);

    scatter1<<<(N_EDGES * 64) / 256, 256, 0, stream>>>(hbuf, ei, agg1);

    gemm2_wmma<<<mblocks, 256, 0, stream>>>(agg1, b1, w2p, hbuf);

    init_out<<<(N_NODES * N_CLASSES + 255) / 256, 256, 0, stream>>>(b2, out);

    scatter2<<<(N_EDGES * 32) / 256, 256, 0, stream>>>(hbuf, ei, out);
}
